// LlamaAttention_52587579572852
// MI455X (gfx1250) — matrix operand order repaired and checked by reading, not compiled, not run on hardware
//
#include <hip/hip_runtime.h>
#include <stdint.h>

#define SEQ    2048
#define DMODEL 2048
#define NHEADS 16
#define HDIM   128
#define NBATCH 2
#define NTOK   (NBATCH * SEQ)

typedef __attribute__((ext_vector_type(16))) __bf16 v16bf;
typedef __attribute__((ext_vector_type(8)))  float  v8f;

union FragAB { v16bf v; uint4 q[2]; };
union U4x8   { uint4 q; unsigned short s[8]; };

__device__ __forceinline__ unsigned short f2bf(float f) {
    unsigned int u = __float_as_uint(f);
    u += 0x7FFFu + ((u >> 16) & 1u);          // round-to-nearest-even
    return (unsigned short)(u >> 16);
}

// Async global -> LDS copy of one 16-byte packet (CDNA5 ASYNCcnt path).
__device__ __forceinline__ void async_b128(const unsigned short* g, unsigned short* lds_generic) {
    unsigned lds = (unsigned)(uintptr_t)lds_generic;   // low 32 bits = LDS offset
    asm volatile("global_load_async_to_lds_b128 %0, %1, off"
                 :: "v"(lds), "v"((unsigned long long)(uintptr_t)g)
                 : "memory");
}
__device__ __forceinline__ void wait_async_le4() {
    asm volatile("s_wait_asynccnt 0x4" ::: "memory");
}
__device__ __forceinline__ void wait_async_0() {
    asm volatile("s_wait_asynccnt 0x0" ::: "memory");
}

// ---------------------------------------------------------------------------
// fp32 -> bf16 conversion (vectorized x4)
// ---------------------------------------------------------------------------
__global__ __launch_bounds__(256)
void cvt_f32_bf16(const float* __restrict__ in, unsigned short* __restrict__ out, int n4) {
    int i = blockIdx.x * blockDim.x + threadIdx.x;
    int stride = gridDim.x * blockDim.x;
    for (; i < n4; i += stride) {
        float4 v = ((const float4*)in)[i];
        ushort4 o;
        o.x = f2bf(v.x); o.y = f2bf(v.y); o.z = f2bf(v.z); o.w = f2bf(v.w);
        ((ushort4*)out)[i] = o;
    }
}

// ---------------------------------------------------------------------------
// Tiled GEMM: out[M][N] = A[M][K](bf16) * W[N][K]^T(bf16) + bias, scaled.
// Block tile 128x128, K-step 32, 256 threads = 8 wave32; wave tile 64x32.
// Double-buffered async global->LDS staging (ASYNCcnt pipeline).
// ---------------------------------------------------------------------------
#define LDT 40   // padded LDS row stride (elements); 80 B keeps 16 B alignment

template <bool OUTF32>
__global__ __launch_bounds__(256)
void gemm_xwt(const unsigned short* __restrict__ A,   // [M][K] bf16, row-major
              const unsigned short* __restrict__ W,   // [N][K] bf16, row-major
              const float* __restrict__ bias,         // [N]
              float alpha,
              unsigned short* __restrict__ outb,      // bf16 out (if !OUTF32)
              float* __restrict__ outf,               // f32 out (if OUTF32)
              int M, int N, int K) {
    __shared__ unsigned short Als[2][128 * LDT];
    __shared__ unsigned short Bls[2][128 * LDT];

    const int wid = threadIdx.x >> 5, lane = threadIdx.x & 31;
    const int wm = wid >> 2, wn = wid & 3;        // 2x4 wave grid
    const int lr = lane & 15, hi = lane >> 4;
    const size_t m0 = (size_t)blockIdx.y * 128;
    const size_t n0 = (size_t)blockIdx.x * 128;

    // Issue one 128x32 tile as 2 async b128 packets per thread (4 per wave
    // including both arrays -> per-wave ASYNCcnt grows by 4 per k-step).
    auto issue_tile = [&](const unsigned short* __restrict__ src, size_t row0,
                          int k0, unsigned short* dst) {
#pragma unroll
        for (int uu = 0; uu < 2; ++uu) {
            const int u  = (int)threadIdx.x + uu * 256;
            const int r  = u >> 2;
            const int c8 = (u & 3) << 3;
            async_b128(src + (row0 + (size_t)r) * (size_t)K + k0 + c8,
                       dst + r * LDT + c8);
        }
    };

    v8f acc[4][2] = {};

    issue_tile(A, m0, 0, Als[0]);
    issue_tile(W, n0, 0, Bls[0]);

    for (int k0 = 0; k0 < K; k0 += 32) {
        const int cur = (k0 >> 5) & 1;
        if (k0 + 32 < K) {
            issue_tile(A, m0, k0 + 32, Als[cur ^ 1]);   // prefetch next tile
            issue_tile(W, n0, k0 + 32, Bls[cur ^ 1]);
            wait_async_le4();                           // wait for current tile only
        } else {
            wait_async_0();
        }
        __syncthreads();

        FragAB af[4], bf2[2];
        const int akb = hi ? 8 : 0;
#pragma unroll
        for (int mf = 0; mf < 4; ++mf) {
            const unsigned short* p = &Als[cur][(wm * 64 + mf * 16 + lr) * LDT + akb];
            af[mf].q[0] = *(const uint4*)p;         // K = akb .. akb+7
            af[mf].q[1] = *(const uint4*)(p + 16);  // K = akb+16 .. akb+23
        }
        const int bkb = hi ? 8 : 0;
#pragma unroll
        for (int nf = 0; nf < 2; ++nf) {
            const unsigned short* p = &Bls[cur][(wn * 32 + nf * 16 + lr) * LDT + bkb];
            bf2[nf].q[0] = *(const uint4*)p;        // K = bkb .. bkb+7
            bf2[nf].q[1] = *(const uint4*)(p + 16); // K = bkb+16 .. bkb+23
        }
#pragma unroll
        for (int mf = 0; mf < 4; ++mf)
#pragma unroll
            for (int nf = 0; nf < 2; ++nf)
                acc[mf][nf] = __builtin_amdgcn_wmma_f32_16x16x32_bf16(
                    false, af[mf].v, false, bf2[nf].v, (short)0, acc[mf][nf],
                    false, false);
        __syncthreads();   // protect buffer reuse by next iteration's asyncs
    }

    // Epilogue: C layout — lane holds col N=lr, rows (hi*8 + r)
#pragma unroll
    for (int mf = 0; mf < 4; ++mf) {
#pragma unroll
        for (int nf = 0; nf < 2; ++nf) {
            const int col = (int)n0 + wn * 32 + nf * 16 + lr;
            const float bv = bias[col];
            const size_t rowbase = m0 + wm * 64 + mf * 16 + hi * 8;
#pragma unroll
            for (int r = 0; r < 8; ++r) {
                float v = (acc[mf][nf][r] + bv) * alpha;
                if (OUTF32) outf[(rowbase + r) * (size_t)N + col] = v;
                else        outb[(rowbase + r) * (size_t)N + col] = f2bf(v);
            }
        }
    }
}

// ---------------------------------------------------------------------------
// Flash attention: block = 64 queries x 1 head x 1 batch; 8 wave32.
// Waves (sm 0..3, sn 0..1): scores tile 16x16 per wave (K-dim 128),
// PV tile 16x64 per wave (K-dim 32 keys). Online softmax via LDS, 4 thr/row.
// K tile staged with async-to-LDS; V transposed through registers.
// ---------------------------------------------------------------------------
#define LDK 136   // K-tile LDS row stride (d contiguous)
#define LDV 40    // transposed-V / P row stride
#define LDS_S 36  // score row stride (floats)

__global__ __launch_bounds__(256)
void flash_attn(const unsigned short* __restrict__ Q,     // [NTOK][D] bf16 (pre-scaled)
                const unsigned short* __restrict__ Kmat,  // [NTOK][D] bf16
                const unsigned short* __restrict__ Vmat,  // [NTOK][D] bf16
                const float* __restrict__ mask,           // [B][S]
                unsigned short* __restrict__ O) {         // [NTOK][D] bf16
    __shared__ unsigned short Ks[32 * LDK];     // [key][d]
    __shared__ unsigned short Vt[HDIM * LDV];   // [d][key]  (transposed)
    __shared__ float          Ss[64 * LDS_S];   // raw scores
    __shared__ unsigned short Ps[64 * LDV];     // exp(scores) bf16
    __shared__ float mstate[64], lstate[64], cstate[64];

    const int wid = threadIdx.x >> 5, lane = threadIdx.x & 31;
    const int sm = wid >> 1, sn = wid & 1;
    const int lr = lane & 15, hi = lane >> 4;
    const size_t q0 = (size_t)blockIdx.x * 64;
    const int h = blockIdx.y, b = blockIdx.z;

    if (threadIdx.x < 64) { mstate[threadIdx.x] = -3.0e38f; lstate[threadIdx.x] = 0.f; }

    // Q fragments held in registers, reused for every key tile
    FragAB qf[4];
    {
        const size_t qrow = (size_t)b * SEQ + q0 + sm * 16 + lr;
        const int akb = hi ? 8 : 0;
#pragma unroll
        for (int kk = 0; kk < 4; ++kk) {
            const unsigned short* p = Q + qrow * DMODEL + h * HDIM + kk * 32 + akb;
            qf[kk].q[0] = *(const uint4*)p;
            qf[kk].q[1] = *(const uint4*)(p + 16);
        }
    }

    v8f oacc[4] = {};

    for (int j0 = 0; j0 < SEQ; j0 += 32) {
        // --- stage K tile via async-to-LDS; V tile transposed via registers ---
        for (int u = threadIdx.x; u < 512; u += 256) {
            const int r  = u >> 4;            // key 0..31
            const int c8 = (u & 15) << 3;     // d base 0..120
            const size_t grow = (size_t)b * SEQ + j0 + r;
            async_b128(Kmat + grow * DMODEL + h * HDIM + c8, &Ks[r * LDK + c8]);
            U4x8 vv; vv.q = *(const uint4*)(Vmat + grow * DMODEL + h * HDIM + c8);
#pragma unroll
            for (int i = 0; i < 8; ++i) Vt[(c8 + i) * LDV + r] = vv.s[i];
        }
        wait_async_0();
        __syncthreads();

        if (j0 + 32 < SEQ) {
            const size_t pr = (size_t)b * SEQ + j0 + 32 + (threadIdx.x & 31);
            __builtin_prefetch(Kmat + pr * DMODEL + h * HDIM, 0, 1);
            __builtin_prefetch(Vmat + pr * DMODEL + h * HDIM, 0, 1);
        }

        // --- scores = Q . K^T  (16q x 16k per wave, head-dim 128) ---
        const int keycol = sn * 16 + lr;
        v8f sc = {};
#pragma unroll
        for (int kk = 0; kk < 4; ++kk) {
            FragAB kf;
            const unsigned short* p = &Ks[keycol * LDK + kk * 32 + hi * 8];
            kf.q[0] = *(const uint4*)p;
            kf.q[1] = *(const uint4*)(p + 16);
            sc = __builtin_amdgcn_wmma_f32_16x16x32_bf16(
                false, qf[kk].v, false, kf.v, (short)0, sc, false, false);
        }
        // additive mask, per key column (matches reference (1-m)*finfo.min)
        {
            const float mv = mask[(size_t)b * SEQ + j0 + keycol];
            const float madd = (1.0f - mv) * -3.4028235e38f;
#pragma unroll
            for (int r = 0; r < 8; ++r) sc[r] += madd;
        }
        {
            const int rowb = sm * 16 + hi * 8;
#pragma unroll
            for (int r = 0; r < 8; ++r) Ss[(rowb + r) * LDS_S + keycol] = sc[r];
        }
        __syncthreads();

        // --- online softmax: 4 threads per row, shfl_xor reductions ---
        {
            const int row = threadIdx.x >> 2;   // 0..63
            const int seg = threadIdx.x & 3;    // 8 columns each
            const float m_old = mstate[row];
            float sval[8];
            float tmax = -3.0e38f;
#pragma unroll
            for (int c = 0; c < 8; ++c) {
                sval[c] = Ss[row * LDS_S + seg * 8 + c];
                tmax = fmaxf(tmax, sval[c]);
            }
            tmax = fmaxf(tmax, __shfl_xor(tmax, 1, 4));
            tmax = fmaxf(tmax, __shfl_xor(tmax, 2, 4));
            const float m_new = fmaxf(m_old, tmax);
            float lsum = 0.f;
            U4x8 pk;
#pragma unroll
            for (int c = 0; c < 8; ++c) {
                const float pv = __expf(sval[c] - m_new);
                lsum += pv;
                pk.s[c] = f2bf(pv);
            }
            *(uint4*)&Ps[row * LDV + seg * 8] = pk.q;   // one ds_store_b128
            lsum += __shfl_xor(lsum, 1, 4);
            lsum += __shfl_xor(lsum, 2, 4);
            if (seg == 0) {
                const float corr = __expf(m_old - m_new);
                mstate[row] = m_new;
                lstate[row] = lstate[row] * corr + lsum;
                cstate[row] = corr;
            }
        }
        __syncthreads();

        // --- O = diag(corr)*O + P.V  (wave: 16 rows x 64 d-cols, K=32 keys) ---
        FragAB pf;
        {
            const int prow = sm * 16 + lr;
            const int akb = hi ? 8 : 0;
            const unsigned short* p = &Ps[prow * LDV + akb];
            pf.q[0] = *(const uint4*)p;
            pf.q[1] = *(const uint4*)(p + 16);
        }
        const int rowb = sm * 16 + hi * 8;
#pragma unroll
        for (int onf = 0; onf < 4; ++onf) {
            const int dcol = sn * 64 + onf * 16 + lr;
            FragAB vf;
            const unsigned short* p = &Vt[dcol * LDV + hi * 8];
            vf.q[0] = *(const uint4*)p;
            vf.q[1] = *(const uint4*)(p + 16);
#pragma unroll
            for (int r = 0; r < 8; ++r) oacc[onf][r] *= cstate[rowb + r];
            oacc[onf] = __builtin_amdgcn_wmma_f32_16x16x32_bf16(
                false, pf.v, false, vf.v, (short)0, oacc[onf], false, false);
        }
        __syncthreads();
    }

    // --- normalize by row sums and store bf16 attn-out (heads concatenated) ---
    const int rowb = sm * 16 + hi * 8;
#pragma unroll
    for (int r = 0; r < 8; ++r) {
        const float invl = 1.0f / lstate[rowb + r];
        const size_t orow = (size_t)b * SEQ + q0 + rowb + r;
#pragma unroll
        for (int onf = 0; onf < 4; ++onf) {
            const int dcol = h * HDIM + sn * 64 + onf * 16 + lr;
            O[orow * DMODEL + dcol] = f2bf(oacc[onf][r] * invl);
        }
    }
}

// ---------------------------------------------------------------------------
extern "C" void kernel_launch(void* const* d_in, const int* in_sizes, int n_in,
                              void* d_out, int out_size, void* d_ws, size_t ws_size,
                              hipStream_t stream) {
    (void)in_sizes; (void)n_in; (void)out_size; (void)ws_size;
    const float* x    = (const float*)d_in[0];
    const float* mask = (const float*)d_in[1];
    const float* Wq   = (const float*)d_in[2];
    const float* bq   = (const float*)d_in[3];
    const float* Wk   = (const float*)d_in[4];
    const float* bk   = (const float*)d_in[5];
    const float* Wv   = (const float*)d_in[6];
    const float* bv   = (const float*)d_in[7];
    const float* Wo   = (const float*)d_in[8];
    const float* bo   = (const float*)d_in[9];
    float* out = (float*)d_out;

    char* ws = (char*)d_ws;
    size_t off = 0;
    auto alloc = [&](size_t bytes) {
        char* p = ws + off;
        off += (bytes + 255) & ~(size_t)255;
        return p;
    };
    unsigned short* Xb  = (unsigned short*)alloc((size_t)NTOK * DMODEL * 2);
    unsigned short* Wqb = (unsigned short*)alloc((size_t)DMODEL * DMODEL * 2);
    unsigned short* Wkb = (unsigned short*)alloc((size_t)DMODEL * DMODEL * 2);
    unsigned short* Wvb = (unsigned short*)alloc((size_t)DMODEL * DMODEL * 2);
    unsigned short* Wob = (unsigned short*)alloc((size_t)DMODEL * DMODEL * 2);
    unsigned short* Qb  = (unsigned short*)alloc((size_t)NTOK * DMODEL * 2);
    unsigned short* Kb  = (unsigned short*)alloc((size_t)NTOK * DMODEL * 2);
    unsigned short* Vb  = (unsigned short*)alloc((size_t)NTOK * DMODEL * 2);
    unsigned short* AOb = (unsigned short*)alloc((size_t)NTOK * DMODEL * 2);

    // 1) down-convert activations + weights to bf16
    auto cvt = [&](const float* src, unsigned short* dst, size_t n) {
        cvt_f32_bf16<<<dim3(512), dim3(256), 0, stream>>>(src, dst, (int)(n / 4));
    };
    cvt(x,  Xb,  (size_t)NTOK * DMODEL);
    cvt(Wq, Wqb, (size_t)DMODEL * DMODEL);
    cvt(Wk, Wkb, (size_t)DMODEL * DMODEL);
    cvt(Wv, Wvb, (size_t)DMODEL * DMODEL);
    cvt(Wo, Wob, (size_t)DMODEL * DMODEL);

    // 2) Q/K/V projections (bias fused; 1/sqrt(head_dim) folded into Q)
    const dim3 gg(DMODEL / 128, NTOK / 128);   // (16, 32)
    const float qscale = 0.08838834764831845f; // 128^-0.5
    gemm_xwt<false><<<gg, 256, 0, stream>>>(Xb, Wqb, bq, qscale, Qb, nullptr,
                                            NTOK, DMODEL, DMODEL);
    gemm_xwt<false><<<gg, 256, 0, stream>>>(Xb, Wkb, bk, 1.0f, Kb, nullptr,
                                            NTOK, DMODEL, DMODEL);
    gemm_xwt<false><<<gg, 256, 0, stream>>>(Xb, Wvb, bv, 1.0f, Vb, nullptr,
                                            NTOK, DMODEL, DMODEL);

    // 3) flash attention (scores never touch HBM)
    flash_attn<<<dim3(SEQ / 64, NHEADS, NBATCH), 256, 0, stream>>>(
        Qb, Kb, Vb, mask, AOb);

    // 4) output projection, fp32 result
    gemm_xwt<true><<<gg, 256, 0, stream>>>(AOb, Wob, bo, 1.0f, nullptr, out,
                                           NTOK, DMODEL, DMODEL);
}
